// CrossModalAttention_58557584113838
// MI455X (gfx1250) — matrix operand order repaired and checked by reading, not compiled, not run on hardware
//
#include <hip/hip_runtime.h>

typedef __bf16 bf16_t;
typedef __attribute__((ext_vector_type(16))) __bf16 v16bf;
typedef __attribute__((ext_vector_type(8)))  __bf16 v8bf;
typedef __attribute__((ext_vector_type(8)))  float  v8f;

#define DEV static __device__ __forceinline__

DEV v8f wmma_bf16(v16bf a, v16bf b, v8f c) {
  // D = A(16x32 bf16) * B(32x16 bf16) + C(16x16 f32)
  return __builtin_amdgcn_wmma_f32_16x16x32_bf16(
      /*neg_a=*/false, a, /*neg_b=*/false, b,
      /*c_mod=*/(short)0, c, /*reuse_a=*/false, /*reuse_b=*/false);
}

// ---- fragment loaders (ISA 7.12.2 layouts) ---------------------------------
// A 16x32 bf16: lanes 0-15 -> M=lane, kbase=0 ; lanes 16-31 -> M=lane-16, kbase=8
// element e: K = kbase + (e&7) + (e>=8 ? 16 : 0)  -> two contiguous 16B halves
DEV v16bf load_a_tile(const bf16_t* base, int ld, int row, int k0, int lane) {
  const int m  = lane & 15;
  const int kb = (lane < 16) ? 0 : 8;
  const bf16_t* p = base + (size_t)(row + m) * ld + (k0 + kb);
  union { v16bf v; v8bf h[2]; } u;
  u.h[0] = *reinterpret_cast<const v8bf*>(p);
  u.h[1] = *reinterpret_cast<const v8bf*>(p + 16);
  return u.v;
}

// Same A fragment, sourced from fp32 and converted (fuses the bf16 cast)
DEV v16bf load_a_tile(const float* base, int ld, int row, int k0, int lane) {
  const int m  = lane & 15;
  const int kb = (lane < 16) ? 0 : 8;
  const float* p = base + (size_t)(row + m) * ld + (k0 + kb);
  v8f x0 = *reinterpret_cast<const v8f*>(p);
  v8f x1 = *reinterpret_cast<const v8f*>(p + 16);
  union { v16bf v; bf16_t e[16]; } u;
#pragma unroll
  for (int i = 0; i < 8; ++i) { u.e[i] = (bf16_t)x0[i]; u.e[8 + i] = (bf16_t)x1[i]; }
  return u.v;
}

// B 32x16 bf16: lane -> output column n=lane&15; same K order as A:
// kbase=0/8 for lanes 0-15/16-31, elements K=kbase+(e&7)+(e>=8?16:0).
// Source memory row = output column, contiguous along K (W[e][k], Vproj[n][k],
// Vt[d][k]).  Two 16B loads per lane.
DEV v16bf load_b_tile(const bf16_t* base, int ld, int col0, int k0, int lane) {
  const int n  = lane & 15;
  const int kb = (lane < 16) ? 0 : 8;
  const bf16_t* p = base + (size_t)(col0 + n) * ld + (k0 + kb);
  union { v16bf v; v8bf h[2]; } u;
  u.h[0] = *reinterpret_cast<const v8bf*>(p);
  u.h[1] = *reinterpret_cast<const v8bf*>(p + 16);
  return u.v;
}

// 16-lane (half-wave) row reductions: xor masks < 16 stay inside each half,
// matching the C/D layout where one matrix row spans lanes 0-15 or 16-31.
DEV float hr_max(float v) {
#pragma unroll
  for (int m = 1; m < 16; m <<= 1) v = fmaxf(v, __shfl_xor(v, m, 32));
  return v;
}
DEV float hr_sum(float v) {
#pragma unroll
  for (int m = 1; m < 16; m <<= 1) v += __shfl_xor(v, m, 32);
  return v;
}

// ---- CDNA5 async global->LDS copy (ASYNCcnt-tracked, no VGPR data path) ----
// Each thread moves 8 x 16B; LDS byte address = low 32 bits of the generic
// pointer (ISA 10.2).
DEV void async_copy_32k(bf16_t* lds_dst, const bf16_t* gsrc, int tid) {
#pragma unroll
  for (int i = 0; i < 8; ++i) {
    const int idx = i * 256 + tid;  // 16-byte units
    const unsigned loff = (unsigned)(size_t)lds_dst + (unsigned)idx * 16u;
    const unsigned long long ga =
        (unsigned long long)(size_t)gsrc + (unsigned long long)idx * 16ull;
    asm volatile("global_load_async_to_lds_b128 %0, %1, off"
                 :: "v"(loff), "v"(ga) : "memory");
  }
}

// ---- GEMM: Y[32768,512] = X[32768,512] * W^T[512,512] + bias ---------------
// W tile (64 output cols x 512 K) is staged fp32->bf16 into 64KB LDS once per
// block and reused by all 8 waves x 16 K-steps.  Each wave owns a 16x64 tile.
template <typename AT, typename OT>
__global__ __launch_bounds__(256)
void gemm512(const AT* __restrict__ X, const float* __restrict__ W,
             const float* __restrict__ bias, OT* __restrict__ Y) {
  __shared__ __align__(32) bf16_t Wl[64 * 512];  // 64 KB

  const int lane = threadIdx.x & 31;
  const int wave = threadIdx.x >> 5;
  const int tid  = threadIdx.x;
  const int row0 = blockIdx.x * 128 + wave * 16;
  const int col0 = blockIdx.y * 64;

  {  // contiguous 128KB fp32 region (rows col0..col0+63 of W) -> bf16 LDS
    const v8f* wsrc = reinterpret_cast<const v8f*>(W + (size_t)col0 * 512);
    v8bf* wdst = reinterpret_cast<v8bf*>(Wl);
    for (int i = tid; i < 4096; i += 256) {
      v8f x = wsrc[i];
      union { v8bf v; bf16_t e[8]; } u;
#pragma unroll
      for (int j = 0; j < 8; ++j) u.e[j] = (bf16_t)x[j];
      wdst[i] = u.v;
    }
  }
  __syncthreads();

  v8f acc[4] = {};
#pragma unroll 4
  for (int s = 0; s < 16; ++s) {
    const v16bf a = load_a_tile(X, 512, row0, s * 32, lane);
#pragma unroll
    for (int j = 0; j < 4; ++j) {
      const v16bf b = load_b_tile(Wl, 512, j * 16, s * 32, lane);
      acc[j] = wmma_bf16(a, b, acc[j]);
    }
  }
  const int c = lane & 15;
  const int h8 = (lane >> 4) * 8;
#pragma unroll
  for (int j = 0; j < 4; ++j) {
    const float bv = bias[col0 + j * 16 + c];
#pragma unroll
    for (int r = 0; r < 8; ++r)
      Y[(size_t)(row0 + h8 + r) * 512 + (col0 + j * 16 + c)] = (OT)(acc[j][r] + bv);
  }
}

// ---- fused flash attention -------------------------------------------------
// Block = 8 waves = 4 row-groups x 2 cooperating waves (64 query rows).
// Within a pair: the K=512 S-reduction is split (each wave holds qa[8]) and
// the d=512 O columns are split (each wave holds 16 accumulator tiles).
// Vproj chunks arrive via double-buffered async global->LDS DMA so the next
// chunk's HBM fetch overlaps the current chunk's 64 WMMAs.
__global__ __launch_bounds__(256)
void flash_attn(const bf16_t* __restrict__ Aproj, const bf16_t* __restrict__ Vproj,
                const float* __restrict__ video, bf16_t* __restrict__ attended) {
  __shared__ __align__(32) bf16_t Vp_lds[2][32 * 512];  // 2 x 32 KB ping-pong
  __shared__ __align__(32) bf16_t Vt_lds[512 * 32];     // 32 KB (raw V, transposed)
  __shared__ __align__(32) bf16_t P_lds[4][16 * 32];    // 4 KB, per-pair P tiles
  __shared__ float Sred[4 * 2 * 2 * 8 * 32];            // 16 KB partial-S exchange

  const int lane = threadIdx.x & 31;
  const int wave = threadIdx.x >> 5;
  const int tid  = threadIdx.x;
  const int pr   = wave >> 1;   // row-group (0..3)
  const int hf   = wave & 1;    // half of K / half of d owned by this wave
  const int bb   = blockIdx.y;
  const int row0 = blockIdx.x * 64 + pr * 16;
  const int c    = lane & 15;
  const int h8   = (lane >> 4) * 8;

  const bf16_t* Qb = Aproj + ((size_t)bb * 2048) * 512;
  const bf16_t* Vb = Vproj + ((size_t)bb * 2048) * 512;
  const float*  Rb = video + ((size_t)bb * 2048) * 512;

  // This wave's half of the Q K-range: steps hf*8 .. hf*8+7 (K = hf*256 ..)
  v16bf qa[8];
#pragma unroll
  for (int j = 0; j < 8; ++j)
    qa[j] = load_a_tile(Qb, 512, row0, (hf * 8 + j) * 32, lane);

  const v8f vzero = {};
  v8f oacc[16];                 // 16 rows x 256 cols fp32 (this wave's half)
#pragma unroll
  for (int t = 0; t < 16; ++t) oacc[t] = vzero;
  float mrow[8], lrow[8];
#pragma unroll
  for (int r = 0; r < 8; ++r) { mrow[r] = -3.0e38f; lrow[r] = 0.0f; }

  const float rs = 0.04419417382415922f;  // 1/sqrt(512)

  // prologue: kick off async DMA of chunk 0 into buffer 0
  async_copy_32k(&Vp_lds[0][0], Vb, tid);

  for (int n0 = 0; n0 < 2048; n0 += 32) {
    const int cur = (n0 >> 5) & 1;
    __syncthreads();  // previous chunk fully consumed (both buffers safe)

    // raw video chunk: vectorized fp32 read, cvt + transposed LDS store
    for (int i = tid; i < 2048; i += 256) {
      const int n = i >> 6, d0 = (i & 63) * 8;
      v8f x = *reinterpret_cast<const v8f*>(Rb + (size_t)(n0 + n) * 512 + d0);
#pragma unroll
      for (int j = 0; j < 8; ++j) Vt_lds[(d0 + j) * 32 + n] = (bf16_t)x[j];
    }
    // issue async DMA for the NEXT Vproj chunk into the other buffer,
    // and prefetch the next raw-video chunk into L2
    if (n0 + 32 < 2048) {
      async_copy_32k(&Vp_lds[cur ^ 1][0], Vb + (size_t)(n0 + 32) * 512, tid);
      const char* pw = (const char*)(Rb + (size_t)(n0 + 32) * 512);
      __builtin_prefetch(pw + (size_t)tid * 256, 0, 1);   // 64 KB fp32
      // current chunk's 8 async loads done once <=8 remain (in-order retire)
      asm volatile("s_wait_asynccnt 0x8" ::: "memory");
    } else {
      asm volatile("s_wait_asynccnt 0x0" ::: "memory");
    }
    __syncthreads();

    // Partial S(16x32) over this wave's K half
    const bf16_t* Vp = &Vp_lds[cur][0];
    v8f s0p = vzero, s1p = vzero;
#pragma unroll
    for (int j = 0; j < 8; ++j) {
      const int k0 = (hf * 8 + j) * 32;
      const v16bf b0 = load_b_tile(Vp, 512, 0,  k0, lane);
      const v16bf b1 = load_b_tile(Vp, 512, 16, k0, lane);
      s0p = wmma_bf16(qa[j], b0, s0p);
      s1p = wmma_bf16(qa[j], b1, s1p);
    }

    // exchange partial S across the pair
#pragma unroll
    for (int r = 0; r < 8; ++r) {
      Sred[(((pr * 2 + hf) * 2 + 0) * 8 + r) * 32 + lane] = s0p[r];
      Sred[(((pr * 2 + hf) * 2 + 1) * 8 + r) * 32 + lane] = s1p[r];
    }
    __syncthreads();

    // combine + online softmax (both waves of a pair compute identical stats)
    float alpha[8], p0v[8], p1v[8];
#pragma unroll
    for (int r = 0; r < 8; ++r) {
      const float a0 =
          (s0p[r] + Sred[(((pr * 2 + (hf ^ 1)) * 2 + 0) * 8 + r) * 32 + lane]) * rs;
      const float a1 =
          (s1p[r] + Sred[(((pr * 2 + (hf ^ 1)) * 2 + 1) * 8 + r) * 32 + lane]) * rs;
      const float cm = hr_max(fmaxf(a0, a1));
      const float mn = fmaxf(mrow[r], cm);
      alpha[r] = __expf(mrow[r] - mn);
      const float p0 = __expf(a0 - mn), p1 = __expf(a1 - mn);
      lrow[r] = lrow[r] * alpha[r] + hr_sum(p0 + p1);
      mrow[r] = mn;
      p0v[r] = p0; p1v[r] = p1;
    }
#pragma unroll
    for (int t = 0; t < 16; ++t) {
#pragma unroll
      for (int r = 0; r < 8; ++r) oacc[t][r] *= alpha[r];
    }

    // C-layout -> A-layout P through LDS; each wave writes its 16-col half
    bf16_t* Pw = &P_lds[pr][0];
#pragma unroll
    for (int r = 0; r < 8; ++r)
      Pw[(h8 + r) * 32 + c + hf * 16] = (bf16_t)(hf ? p1v[r] : p0v[r]);
    __syncthreads();
    const v16bf pa = load_a_tile((const bf16_t*)Pw, 32, 0, 0, lane);

    // O(16 x 256) += P(16x32) * Vraw_chunk(32 x this wave's d-half)
#pragma unroll
    for (int t = 0; t < 16; ++t) {
      const int tt = hf * 16 + t;
      const v16bf bv = load_b_tile(Vt_lds, 32, tt * 16, 0, lane);
      oacc[t] = wmma_bf16(pa, bv, oacc[t]);
    }
  }

  // normalize + store this wave's half of attended (bf16)
  float inv[8];
#pragma unroll
  for (int r = 0; r < 8; ++r) inv[r] = 1.0f / lrow[r];
  bf16_t* Ob = attended + ((size_t)bb * 2048 + row0) * 512;
#pragma unroll
  for (int t = 0; t < 16; ++t) {
    const int colb = (hf * 16 + t) * 16 + c;
#pragma unroll
    for (int r = 0; r < 8; ++r)
      Ob[(size_t)(h8 + r) * 512 + colb] = (bf16_t)(oacc[t][r] * inv[r]);
  }
}

// ---- host-side launch ------------------------------------------------------
extern "C" void kernel_launch(void* const* d_in, const int* in_sizes, int n_in,
                              void* d_out, int out_size, void* d_ws, size_t ws_size,
                              hipStream_t stream) {
  (void)in_sizes; (void)n_in; (void)out_size; (void)ws_size;
  const float* audio = (const float*)d_in[0];
  const float* video = (const float*)d_in[1];
  const float* W_a   = (const float*)d_in[2];
  const float* b_a   = (const float*)d_in[3];
  const float* W_v   = (const float*)d_in[4];
  const float* b_v   = (const float*)d_in[5];
  const float* W_o   = (const float*)d_in[6];
  const float* b_o   = (const float*)d_in[7];
  float* out = (float*)d_out;

  char* ws = (char*)d_ws;
  const size_t PSZ = (size_t)16 * 2048 * 512 * sizeof(bf16_t);  // 32 MB
  bf16_t* Aproj = (bf16_t*)(ws);
  bf16_t* Vproj = (bf16_t*)(ws + PSZ);
  bf16_t* Attn  = (bf16_t*)(ws + 2 * PSZ);                      // 96 MB total

  dim3 g1(32768 / 128, 512 / 64);   // 256 x 8 blocks
  gemm512<float, bf16_t><<<g1, 256, 0, stream>>>(audio, W_a, b_a, Aproj);
  gemm512<float, bf16_t><<<g1, 256, 0, stream>>>(video, W_v, b_v, Vproj);

  dim3 g2(2048 / 64, 16);           // 32 row-tiles x 16 batches
  flash_attn<<<g2, 256, 0, stream>>>(Aproj, Vproj, video, Attn);

  gemm512<bf16_t, float><<<g1, 256, 0, stream>>>(Attn, W_o, b_o, out);
}
